// DirectionAwareMessagePassing_25933012533743
// MI455X (gfx1250) — matrix operand order repaired and checked by reading, not compiled, not run on hardware
//
#include <hip/hip_runtime.h>
#include <hip/hip_bf16.h>

// ---------------- problem constants ----------------
#define BB   8
#define NN   128
#define RR   4096
#define DD   1024
#define EE   200
#define CC   151
#define PIN  9
#define CATK 1352          // D + E + 128
#define CATKP 1376         // padded to multiple of 32
#define CPAD 192           // classes padded to multiple of 64 (GEMM N-tile)
#define ROWS (BB*NN)       // 1024

typedef __attribute__((ext_vector_type(16))) __bf16 v16bf;
typedef __attribute__((ext_vector_type(8)))  float  v8f;

union FragB16 { v16bf v; uint4 u4[2]; };
static_assert(sizeof(v16bf) == 32, "v16bf must be 32 bytes");

// ---- fragment loaders (CDNA5 wave32 WMMA 16x16x32 bf16 layouts) ----
// A (16x32, row-major source): lane l holds row (l&15);
//   lanes 0-15 : K = k0+{0..7} and k0+{16..23}
//   lanes 16-31: K = k0+{8..15} and k0+{24..31}
__device__ inline v16bf fragA_bf16(const __bf16* __restrict__ row, int k0, int h) {
  FragB16 f;
  f.u4[0] = *(const uint4*)(row + k0 + 8*h);
  f.u4[1] = *(const uint4*)(row + k0 + 8*h + 16);
  return f.v;
}
// B (32x16) from weight stored transposed [N][Kpad]: lane l holds column (l&15);
//   lanes 0-15: K = k0+{0..7},{16..23} ; lanes 16-31: K = k0+{8..15},{24..31}
__device__ inline v16bf fragB_bf16(const __bf16* __restrict__ colptr, int k0, int h) {
  FragB16 f;
  f.u4[0] = *(const uint4*)(colptr + k0 + 8*h);
  f.u4[1] = *(const uint4*)(colptr + k0 + 8*h + 16);
  return f.v;
}

// ---------------- generic batched WMMA GEMM (16x64 tile / wave) ----------------
// C[b][m][n] = sum_k A[b][m][k] * Bt[b][n][k]  (+bias[n]) (opt ReLU)
// One wave computes a 16x64 tile: A fragment reused across 4 WMMAs.
// K % 32 == 0, M % 16 == 0, N % 64 == 0.
__global__ void __launch_bounds__(32)
k_gemm(const __bf16* __restrict__ A,  long long sA, int lda,
       const __bf16* __restrict__ Bt, long long sB, int ldb,
       float* __restrict__ C,         long long sC, int ldc,
       const float* __restrict__ bias, int relu, int K)
{
  const int tn = blockIdx.x, tm = blockIdx.y, b = blockIdx.z;
  const int lane = threadIdx.x;
  const int h = lane >> 4, c15 = lane & 15;
  const __bf16* Ab = A + (size_t)b*sA + (size_t)(tm*16 + c15)*lda;
  const __bf16* Bb0 = Bt + (size_t)b*sB + (size_t)(tn*64 +  0 + c15)*ldb;
  const __bf16* Bb1 = Bt + (size_t)b*sB + (size_t)(tn*64 + 16 + c15)*ldb;
  const __bf16* Bb2 = Bt + (size_t)b*sB + (size_t)(tn*64 + 32 + c15)*ldb;
  const __bf16* Bb3 = Bt + (size_t)b*sB + (size_t)(tn*64 + 48 + c15)*ldb;
  v8f acc0 = {0.f,0.f,0.f,0.f,0.f,0.f,0.f,0.f};
  v8f acc1 = acc0, acc2 = acc0, acc3 = acc0;
  for (int k0 = 0; k0 < K; k0 += 32) {
    __builtin_prefetch(Ab  + k0 + 256, 0, 1);
    __builtin_prefetch(Bb0 + k0 + 256, 0, 1);
    const v16bf a  = fragA_bf16(Ab, k0, h);
    const v16bf b0 = fragB_bf16(Bb0, k0, h);
    const v16bf b1 = fragB_bf16(Bb1, k0, h);
    const v16bf b2 = fragB_bf16(Bb2, k0, h);
    const v16bf b3 = fragB_bf16(Bb3, k0, h);
    acc0 = __builtin_amdgcn_wmma_f32_16x16x32_bf16(false, a, false, b0, (short)0, acc0, false, false);
    acc1 = __builtin_amdgcn_wmma_f32_16x16x32_bf16(false, a, false, b1, (short)0, acc1, false, false);
    acc2 = __builtin_amdgcn_wmma_f32_16x16x32_bf16(false, a, false, b2, (short)0, acc2, false, false);
    acc3 = __builtin_amdgcn_wmma_f32_16x16x32_bf16(false, a, false, b3, (short)0, acc3, false, false);
  }
  float* Cp = C + (size_t)b*sC;
  v8f accs[4] = {acc0, acc1, acc2, acc3};
#pragma unroll
  for (int j = 0; j < 4; ++j) {
    const int col = tn*64 + 16*j + c15;
    const float bvv = bias ? bias[col] : 0.f;
#pragma unroll
    for (int v = 0; v < 8; ++v) {
      const int rowm = tm*16 + v + 8*h;
      float val = accs[j][v] + bvv;
      if (relu) val = fmaxf(val, 0.f);
      Cp[(size_t)rowm*ldc + col] = val;
    }
  }
}

// ---------------- fused s/o/u attention-coefficient kernel ----------------
// For a tile of 16 pairs: c_r = sum_d (s+bs)(o+bo)(u+bu)*w_w[d] + w_b,
// with s = rep[subj]@Ws, o = rep[obj]@Wo, u = union@Wu  (bf16 WMMA, f32 acc).
// Each wave processes 2 d-blocks per A-fragment load (6 WMMA / A-triple).
// Scatter-add c_r into dense Amat[b][subj][obj].
__global__ void __launch_bounds__(128)
k_coeff(const __bf16* __restrict__ repbf, const __bf16* __restrict__ unibf,
        const int* __restrict__ pairs,
        const __bf16* __restrict__ WsT, const __bf16* __restrict__ WoT,
        const __bf16* __restrict__ WuT,
        const float* __restrict__ ws_b, const float* __restrict__ wo_b,
        const float* __restrict__ wu_b,
        const float* __restrict__ w_w,  const float* __restrict__ w_b,
        float* __restrict__ Amat)
{
  const int b  = blockIdx.y;
  const int r0 = blockIdx.x * 16;
  const int t  = threadIdx.x;
  const int wv = t >> 5, lane = t & 31;
  const int h = lane >> 4, c15 = lane & 15;

  __shared__ float coeffAcc[16];
  if (t < 16) coeffAcc[t] = 0.f;
  __syncthreads();

  const int r    = r0 + c15;                       // A-fragment row for this lane
  const int subj = pairs[((size_t)b*RR + r)*2 + 0];
  const int obj  = pairs[((size_t)b*RR + r)*2 + 1];
  const __bf16* rs = repbf + ((size_t)b*NN + subj)*DD;
  const __bf16* ro = repbf + ((size_t)b*NN + obj )*DD;
  const __bf16* ru = unibf + ((size_t)b*RR + r   )*DD;

  float acc[8];
#pragma unroll
  for (int v = 0; v < 8; ++v) acc[v] = 0.f;

  for (int nbp = wv; nbp < DD/32; nbp += 4) {      // 8 iterations, 2 d-blocks each
    const int d0 = nbp * 32;
    const __bf16* Bs0 = WsT + (size_t)(d0 +      c15)*DD;
    const __bf16* Bs1 = WsT + (size_t)(d0 + 16 + c15)*DD;
    const __bf16* Bo0 = WoT + (size_t)(d0 +      c15)*DD;
    const __bf16* Bo1 = WoT + (size_t)(d0 + 16 + c15)*DD;
    const __bf16* Bu0 = WuT + (size_t)(d0 +      c15)*DD;
    const __bf16* Bu1 = WuT + (size_t)(d0 + 16 + c15)*DD;
    v8f cs0 = {0.f,0.f,0.f,0.f,0.f,0.f,0.f,0.f};
    v8f cs1 = cs0, co0 = cs0, co1 = cs0, cu0 = cs0, cu1 = cs0;
    for (int k0 = 0; k0 < DD; k0 += 32) {
      const v16bf as = fragA_bf16(rs, k0, h);
      const v16bf ao = fragA_bf16(ro, k0, h);
      const v16bf au = fragA_bf16(ru, k0, h);
      cs0 = __builtin_amdgcn_wmma_f32_16x16x32_bf16(false, as, false, fragB_bf16(Bs0, k0, h), (short)0, cs0, false, false);
      cs1 = __builtin_amdgcn_wmma_f32_16x16x32_bf16(false, as, false, fragB_bf16(Bs1, k0, h), (short)0, cs1, false, false);
      co0 = __builtin_amdgcn_wmma_f32_16x16x32_bf16(false, ao, false, fragB_bf16(Bo0, k0, h), (short)0, co0, false, false);
      co1 = __builtin_amdgcn_wmma_f32_16x16x32_bf16(false, ao, false, fragB_bf16(Bo1, k0, h), (short)0, co1, false, false);
      cu0 = __builtin_amdgcn_wmma_f32_16x16x32_bf16(false, au, false, fragB_bf16(Bu0, k0, h), (short)0, cu0, false, false);
      cu1 = __builtin_amdgcn_wmma_f32_16x16x32_bf16(false, au, false, fragB_bf16(Bu1, k0, h), (short)0, cu1, false, false);
    }
    const int da = d0 + c15, db = d0 + 16 + c15;   // output columns for this lane
    const float sba = ws_b[da], oba = wo_b[da], uba = wu_b[da], wwa = w_w[da];
    const float sbb = ws_b[db], obb = wo_b[db], ubb = wu_b[db], wwb = w_w[db];
#pragma unroll
    for (int v = 0; v < 8; ++v) {
      acc[v] += (cs0[v] + sba) * (co0[v] + oba) * (cu0[v] + uba) * wwa
              + (cs1[v] + sbb) * (co1[v] + obb) * (cu1[v] + ubb) * wwb;
    }
  }
#pragma unroll
  for (int v = 0; v < 8; ++v)
    atomicAdd(&coeffAcc[v + 8*h], acc[v]);         // pair row = v + 8*(lane>>4)
  __syncthreads();
  if (t < 16) {
    const int rr = r0 + t;
    const int s2 = pairs[((size_t)b*RR + rr)*2 + 0];
    const int o2 = pairs[((size_t)b*RR + rr)*2 + 1];
    atomicAdd(&Amat[((size_t)b*NN + s2)*NN + o2], coeffAcc[t] + w_b[0]);
  }
}

// ---------------- small VALU kernels ----------------
__global__ void k_convert_wT(const float* __restrict__ W, __bf16* __restrict__ Wt,
                             int K, int N, int Kpad, int Npad)
{
  const int total = Npad * Kpad;
  for (int i = blockIdx.x*blockDim.x + threadIdx.x; i < total; i += gridDim.x*blockDim.x) {
    const int n = i / Kpad, k = i % Kpad;
    const float v = (k < K && n < N) ? W[(size_t)k*N + n] : 0.f;
    Wt[i] = (__bf16)v;
  }
}

__global__ void k_f32_to_bf16(const float* __restrict__ s, __bf16* __restrict__ d, int n) {
  for (int i = blockIdx.x*blockDim.x + threadIdx.x; i < n; i += gridDim.x*blockDim.x)
    d[i] = (__bf16)s[i];
}

__global__ void k_zero(float* __restrict__ p, int n) {
  for (int i = blockIdx.x*blockDim.x + threadIdx.x; i < n; i += gridDim.x*blockDim.x)
    p[i] = 0.f;
}

__global__ void k_padbias(const float* __restrict__ b, float* __restrict__ bp) {
  const int i = threadIdx.x;
  if (i < CPAD) bp[i] = (i < CC) ? b[i] : 0.f;
}

// front-end: obj_embed + pos_embed + concat -> cat_bf [ROWS][CATKP]
__global__ void __launch_bounds__(256)
k_cat(const float* __restrict__ obj_feats, const float* __restrict__ box_info,
      const float* __restrict__ obj_dists, const float* __restrict__ embed_w,
      const float* __restrict__ pe_w1, const float* __restrict__ pe_b1,
      const float* __restrict__ pe_g,  const float* __restrict__ pe_be,
      const float* __restrict__ pe_w2, const float* __restrict__ pe_b2,
      __bf16* __restrict__ cat)
{
  const int row = blockIdx.x, t = threadIdx.x;
  __shared__ float sb[PIN];
  __shared__ float sh[32];
  __shared__ float sd[CC + 1];
  if (t < PIN) sb[t] = box_info[(size_t)row*PIN + t];
  if (t < CC)  sd[t] = obj_dists[(size_t)row*CC + t];
  __syncthreads();
  if (t < 32) {
    float hv = pe_b1[t];
    for (int i = 0; i < PIN; ++i) hv += sb[i]*pe_w1[i*32 + t];
    const float invs = 0.999995000037f;            // 1/sqrt(1+1e-5)  (BN eval)
    sh[t] = hv*invs*pe_g[t] + pe_be[t];
  }
  __syncthreads();
  __bf16* crow = cat + (size_t)row*CATKP;
  for (int c = t; c < DD; c += 256) crow[c] = (__bf16)obj_feats[(size_t)row*DD + c];
  if (t < EE) {
    float e = 0.f;
    for (int i = 0; i < CC; ++i) e += sd[i]*embed_w[i*EE + t];
    crow[DD + t] = (__bf16)e;
  }
  if (t < 128) {
    float p = pe_b2[t];
    for (int i = 0; i < 32; ++i) p += sh[i]*pe_w2[i*128 + t];
    crow[DD + EE + t] = (__bf16)fmaxf(p, 0.f);
  }
  if (t >= EE && t < EE + (CATKP - CATK)) crow[CATK + (t - EE)] = (__bf16)0.f;
}

// sigmoid, zero-diagonal, row-normalize; emit bf16 A and A^T
__global__ void __launch_bounds__(NN)
k_afinal(const float* __restrict__ Amat, __bf16* __restrict__ Abf, __bf16* __restrict__ ATbf)
{
  const int b = blockIdx.y, n = blockIdx.x, m = threadIdx.x;
  __shared__ float red[NN];
  float v = Amat[((size_t)b*NN + n)*NN + m];
  v = 1.f / (1.f + __expf(-v));
  if (m == n) v = 0.f;
  red[m] = v; __syncthreads();
  for (int s = NN/2; s > 0; s >>= 1) { if (m < s) red[m] += red[m + s]; __syncthreads(); }
  const float a = v / red[0];
  Abf [((size_t)b*NN + n)*NN + m] = (__bf16)a;
  ATbf[((size_t)b*NN + m)*NN + n] = (__bf16)a;
}

// W3 [ROWS][512] f32 -> per-image transposed bf16 [b][512][NN]
__global__ void k_w3t(const float* __restrict__ W3, __bf16* __restrict__ W3T) {
  const int total = BB*512*NN;
  for (int i = blockIdx.x*blockDim.x + threadIdx.x; i < total; i += gridDim.x*blockDim.x) {
    const int b = i / (512*NN), rem = i % (512*NN);
    const int d = rem / NN, n = rem % NN;
    W3T[i] = (__bf16)W3[((size_t)b*NN + n)*512 + d];
  }
}

// LayerNorm(256) * g + be, ReLU -> bf16
__global__ void __launch_bounds__(256)
k_ln(const float* __restrict__ t, const float* __restrict__ g,
     const float* __restrict__ be, __bf16* __restrict__ t2)
{
  const int row = blockIdx.x, i = threadIdx.x;
  __shared__ float red[256];
  const float x = t[(size_t)row*256 + i];
  red[i] = x; __syncthreads();
  for (int s = 128; s > 0; s >>= 1) { if (i < s) red[i] += red[i + s]; __syncthreads(); }
  const float mu = red[0] / 256.f;
  __syncthreads();
  const float d = x - mu;
  red[i] = d*d; __syncthreads();
  for (int s = 128; s > 0; s >>= 1) { if (i < s) red[i] += red[i + s]; __syncthreads(); }
  const float var = red[0] / 256.f;
  const float y = d * rsqrtf(var + 1e-5f) * g[i] + be[i];
  t2[(size_t)row*256 + i] = (__bf16)fmaxf(y, 0.f);
}

// ctx_rep = relu(rep + nb) -> f32 (d_out region) + bf16 copy
__global__ void k_residual(const float* __restrict__ rep, const float* __restrict__ nb,
                           float* __restrict__ outCtx, __bf16* __restrict__ crepbf)
{
  const int n = ROWS*DD;
  for (int i = blockIdx.x*blockDim.x + threadIdx.x; i < n; i += gridDim.x*blockDim.x) {
    const float v = fmaxf(rep[i] + nb[i], 0.f);
    outCtx[i] = v;
    crepbf[i] = (__bf16)v;
  }
}

// copy scores, argmax over classes 1..150 (+1), first-max tie-break
__global__ void __launch_bounds__(256)
k_scores(const float* __restrict__ spad, float* __restrict__ outScores,
         float* __restrict__ outPreds)
{
  const int row = blockIdx.x, t = threadIdx.x;
  __shared__ float sc[CC];
  if (t < CC) {
    const float v = spad[(size_t)row*CPAD + t];
    sc[t] = v;
    outScores[(size_t)row*CC + t] = v;
  }
  __syncthreads();
  if (t == 0) {
    int best = 1; float bv = sc[1];
    for (int c = 2; c < CC; ++c) if (sc[c] > bv) { bv = sc[c]; best = c; }
    outPreds[row] = (float)best;
  }
}

// ---------------- host launcher ----------------
extern "C" void kernel_launch(void* const* d_in, const int* in_sizes, int n_in,
                              void* d_out, int out_size, void* d_ws, size_t ws_size,
                              hipStream_t stream)
{
  (void)in_sizes; (void)n_in; (void)out_size; (void)ws_size;
  const float* obj_feats = (const float*)d_in[0];
  const float* union_f   = (const float*)d_in[1];
  const float* box_info  = (const float*)d_in[2];
  const float* obj_dists = (const float*)d_in[3];
  const int*   pairs     = (const int*)  d_in[4];
  const float* embed_w   = (const float*)d_in[5];
  const float* pe_w1 = (const float*)d_in[6];  const float* pe_b1 = (const float*)d_in[7];
  const float* pe_g  = (const float*)d_in[8];  const float* pe_be = (const float*)d_in[9];
  const float* pe_w2 = (const float*)d_in[10]; const float* pe_b2 = (const float*)d_in[11];
  const float* proj_w = (const float*)d_in[12]; const float* proj_b = (const float*)d_in[13];
  const float* ws_w = (const float*)d_in[14]; const float* ws_b = (const float*)d_in[15];
  const float* wo_w = (const float*)d_in[16]; const float* wo_b = (const float*)d_in[17];
  const float* wu_w = (const float*)d_in[18]; const float* wu_b = (const float*)d_in[19];
  const float* w_w  = (const float*)d_in[20]; const float* w_b  = (const float*)d_in[21];
  const float* wt3_w = (const float*)d_in[22]; const float* wt3_b = (const float*)d_in[23];
  const float* tr_w1 = (const float*)d_in[24]; const float* tr_b1 = (const float*)d_in[25];
  const float* tr_g  = (const float*)d_in[26]; const float* tr_be = (const float*)d_in[27];
  const float* tr_w2 = (const float*)d_in[28]; const float* tr_b2 = (const float*)d_in[29];
  const float* out_w = (const float*)d_in[30]; const float* out_b = (const float*)d_in[31];

  // workspace layout (256B aligned)
  char* ws = (char*)d_ws;
  size_t off = 0;
  auto nxt = [&off](size_t bytes) { size_t r = off; off = (off + bytes + 255) & ~(size_t)255; return r; };
  __bf16* wsT   = (__bf16*)(ws + nxt((size_t)DD*DD*2));
  __bf16* woT   = (__bf16*)(ws + nxt((size_t)DD*DD*2));
  __bf16* wuT   = (__bf16*)(ws + nxt((size_t)DD*DD*2));
  __bf16* projT = (__bf16*)(ws + nxt((size_t)DD*CATKP*2));
  __bf16* wt3T  = (__bf16*)(ws + nxt((size_t)512*DD*2));
  __bf16* tr1T  = (__bf16*)(ws + nxt((size_t)256*DD*2));
  __bf16* tr2T  = (__bf16*)(ws + nxt((size_t)DD*256*2));
  __bf16* outT  = (__bf16*)(ws + nxt((size_t)CPAD*DD*2));
  float*  outbp = (float*) (ws + nxt((size_t)CPAD*4));
  __bf16* catb  = (__bf16*)(ws + nxt((size_t)ROWS*CATKP*2));
  float*  rep   = (float*) (ws + nxt((size_t)ROWS*DD*4));
  __bf16* repbf = (__bf16*)(ws + nxt((size_t)ROWS*DD*2));
  __bf16* unibf = (__bf16*)(ws + nxt((size_t)BB*RR*DD*2));   // 64 MB
  float*  Amat  = (float*) (ws + nxt((size_t)BB*NN*NN*4));
  __bf16* Abf   = (__bf16*)(ws + nxt((size_t)BB*NN*NN*2));
  __bf16* ATbf  = (__bf16*)(ws + nxt((size_t)BB*NN*NN*2));
  float*  W3    = (float*) (ws + nxt((size_t)ROWS*512*4));
  __bf16* W3T   = (__bf16*)(ws + nxt((size_t)BB*512*NN*2));
  float*  ctx   = (float*) (ws + nxt((size_t)ROWS*DD*4));
  __bf16* ctxbf = (__bf16*)(ws + nxt((size_t)ROWS*DD*2));
  float*  tbuf  = (float*) (ws + nxt((size_t)ROWS*256*4));
  __bf16* t2bf  = (__bf16*)(ws + nxt((size_t)ROWS*256*2));
  float*  nbuf  = (float*) (ws + nxt((size_t)ROWS*DD*4));
  __bf16* crepbf= (__bf16*)(ws + nxt((size_t)ROWS*DD*2));
  float*  spad  = (float*) (ws + nxt((size_t)ROWS*CPAD*4));

  float* outScores = (float*)d_out;                 // [ROWS*CC]
  float* outPreds  = outScores + (size_t)ROWS*CC;   // [ROWS]
  float* outCtx    = outPreds + ROWS;               // [ROWS*DD]

  auto grd = [](long long n) { return dim3((unsigned)((n + 255) / 256)); };

  // 1) weight conversions (transposed bf16, zero-padded)
  k_convert_wT<<<grd(DD*DD),      256, 0, stream>>>(ws_w,  wsT,  DD,   DD,  DD,    DD);
  k_convert_wT<<<grd(DD*DD),      256, 0, stream>>>(wo_w,  woT,  DD,   DD,  DD,    DD);
  k_convert_wT<<<grd(DD*DD),      256, 0, stream>>>(wu_w,  wuT,  DD,   DD,  DD,    DD);
  k_convert_wT<<<grd(DD*CATKP),   256, 0, stream>>>(proj_w,projT,CATK, DD,  CATKP, DD);
  k_convert_wT<<<grd(512*DD),     256, 0, stream>>>(wt3_w, wt3T, DD,   512, DD,    512);
  k_convert_wT<<<grd(256*DD),     256, 0, stream>>>(tr_w1, tr1T, DD,   256, DD,    256);
  k_convert_wT<<<grd(DD*256),     256, 0, stream>>>(tr_w2, tr2T, 256,  DD,  256,   DD);
  k_convert_wT<<<grd(CPAD*DD),    256, 0, stream>>>(out_w, outT, DD,   CC,  DD,    CPAD);
  k_padbias<<<1, 256, 0, stream>>>(out_b, outbp);

  // 2) activation conversions + front-end concat
  k_f32_to_bf16<<<grd((long long)BB*RR*DD), 256, 0, stream>>>(union_f, unibf, BB*RR*DD);
  k_cat<<<ROWS, 256, 0, stream>>>(obj_feats, box_info, obj_dists, embed_w,
                                  pe_w1, pe_b1, pe_g, pe_be, pe_w2, pe_b2, catb);

  // 3) rep = relu(cat @ proj_w + proj_b)   [ROWS x 1024]
  k_gemm<<<dim3(DD/64, ROWS/16, 1), 32, 0, stream>>>(
      catb, 0, CATKP, projT, 0, CATKP, rep, 0, DD, proj_b, 1, CATKP);
  k_f32_to_bf16<<<grd(ROWS*DD), 256, 0, stream>>>(rep, repbf, ROWS*DD);

  // 4) fused direction-aware coefficients -> dense Amat (scatter-add)
  k_zero<<<grd(BB*NN*NN), 256, 0, stream>>>(Amat, BB*NN*NN);
  k_coeff<<<dim3(RR/16, BB), 128, 0, stream>>>(repbf, unibf, pairs,
      wsT, woT, wuT, ws_b, wo_b, wu_b, w_w, w_b, Amat);

  // 5) sigmoid, mask diagonal, row-normalize; bf16 A and A^T
  k_afinal<<<dim3(NN, BB), NN, 0, stream>>>(Amat, Abf, ATbf);

  // 6) W3 = relu(rep @ wt3_w + wt3_b)   [ROWS x 512], per-image transpose
  k_gemm<<<dim3(512/64, ROWS/16, 1), 32, 0, stream>>>(
      repbf, 0, DD, wt3T, 0, DD, W3, 0, 512, wt3_b, 1, DD);
  k_w3t<<<grd(BB*512*NN), 256, 0, stream>>>(W3, W3T);

  // 7) ctx = [A@W3 | A^T@W3]  (batched over images, K=128)
  k_gemm<<<dim3(512/64, NN/16, BB), 32, 0, stream>>>(
      Abf,  (long long)NN*NN, NN, W3T, (long long)512*NN, NN,
      ctx,       (long long)NN*DD, DD, nullptr, 0, NN);
  k_gemm<<<dim3(512/64, NN/16, BB), 32, 0, stream>>>(
      ATbf, (long long)NN*NN, NN, W3T, (long long)512*NN, NN,
      ctx + 512, (long long)NN*DD, DD, nullptr, 0, NN);
  k_f32_to_bf16<<<grd(ROWS*DD), 256, 0, stream>>>(ctx, ctxbf, ROWS*DD);

  // 8) t = ctx @ tr_w1 + tr_b1 ; LayerNorm ; ReLU
  k_gemm<<<dim3(256/64, ROWS/16, 1), 32, 0, stream>>>(
      ctxbf, 0, DD, tr1T, 0, DD, tbuf, 0, 256, tr_b1, 0, DD);
  k_ln<<<ROWS, 256, 0, stream>>>(tbuf, tr_g, tr_be, t2bf);

  // 9) nb = relu(t) @ tr_w2 + tr_b2 ; ctx_rep = relu(rep + nb)
  k_gemm<<<dim3(DD/64, ROWS/16, 1), 32, 0, stream>>>(
      t2bf, 0, 256, tr2T, 0, 256, nbuf, 0, DD, tr_b2, 0, 256);
  k_residual<<<grd(ROWS*DD), 256, 0, stream>>>(rep, nbuf, outCtx, crepbf);

  // 10) scores = ctx_rep @ out_w + out_b ; preds = argmax(scores[:,1:]) + 1
  k_gemm<<<dim3(CPAD/64, ROWS/16, 1), 32, 0, stream>>>(
      crepbf, 0, DD, outT, 0, DD, spad, 0, CPAD, outbp, 0, DD);
  k_scores<<<ROWS, 256, 0, stream>>>(spad, outScores, outPreds);
}
